// LSTMNet_23141283791162
// MI455X (gfx1250) — matrix operand order repaired and checked by reading, not compiled, not run on hardware
//
#include <hip/hip_runtime.h>
#include <cstdint>
#include <cstddef>

// ---------- types ----------
typedef __attribute__((ext_vector_type(16))) __bf16   v16bf;
typedef __attribute__((ext_vector_type(8)))  float    v8f;
typedef __attribute__((ext_vector_type(4)))  unsigned u32x4;
typedef __attribute__((ext_vector_type(4)))  float    f32x4;
typedef __attribute__((ext_vector_type(2)))  unsigned long long u64x2;
typedef __attribute__((ext_vector_type(4)))  __bf16   v4bf;
typedef int v4i_async __attribute__((vector_size(16)));   // matches async-LDS builtin param

union BF16Frag { v16bf v; u32x4 u[2]; };

// ---------- problem sizes ----------
#define TT   512
#define BB   64
#define II   256
#define HH   512
#define OO   1024
#define G4H  2048
#define NBLK 32      // one workgroup per 16-wide H slice
#define HPAD 520     // padded LDS row stride (bf16 elems): 1040B rows, 16B aligned
#define GPAD 20      // padded gates row stride (f32): 80B rows, 16B aligned

// CDNA5 async load-to-LDS path (ASYNCcnt-tracked), if this toolchain exposes it.
#if defined(__has_builtin)
#if __has_builtin(__builtin_amdgcn_global_load_async_to_lds_b128) && \
    __has_builtin(__builtin_amdgcn_s_wait_asynccnt)
#define USE_ASYNC_LDS 1
#endif
#endif

__device__ __forceinline__ float sigmoidf_(float x) {
  return 1.0f / (1.0f + __expf(-x));
}

// ---------- f32 -> bf16 conversion ----------
__global__ __launch_bounds__(256) void cvt_bf16_kernel(const float* __restrict__ src,
                                                       __bf16* __restrict__ dst, int n) {
  int i = (blockIdx.x * 256 + threadIdx.x) * 4;
  if (i < n) {
    f32x4 f = *(const f32x4*)&src[i];
    v4bf o;
    o.x = (__bf16)f.x; o.y = (__bf16)f.y; o.z = (__bf16)f.z; o.w = (__bf16)f.w;
    *(v4bf*)&dst[i] = o;
  }
}

// ---------- zero h(0), reset grid barrier ----------
__global__ __launch_bounds__(256) void init_kernel(unsigned* hs0, unsigned* bar) {
  int i = blockIdx.x * 256 + threadIdx.x;
  if (i < (BB * HH) / 2) hs0[i] = 0u;   // 16384 dwords of bf16 zeros
  if (i == 0) *bar = 0u;
}

// ---------- generic bf16 WMMA GEMM: C(MxN) = A(MxK) * W(NxK)^T + bias ----------
// block = 256 thr = 8 waves, block tile 128x128, wave tile 64x32 (4x2 frags)
__global__ __launch_bounds__(256) void gemm_bf16_kernel(
    const __bf16* __restrict__ A, const __bf16* __restrict__ W,
    const float* __restrict__ bias0, const float* __restrict__ bias1,
    float* __restrict__ C, int M, int N, int K)
{
  const int lane = threadIdx.x & 31;
  const int w    = threadIdx.x >> 5;
  const int wy   = w & 1;
  const int wx   = w >> 1;
  const int m_base = blockIdx.y * 128 + wy * 64;
  const int n_base = blockIdx.x * 128 + wx * 32;
  const int mrow = lane & 15;
  const int ksel = lane >> 4;

  v8f acc[4][2] = {};

  const __bf16* Ap[4];
  #pragma unroll
  for (int mt = 0; mt < 4; ++mt)
    Ap[mt] = A + (size_t)(m_base + mt * 16 + mrow) * K + ksel * 8;
  const __bf16* Wp[2];
  #pragma unroll
  for (int nt = 0; nt < 2; ++nt)
    Wp[nt] = W + (size_t)(n_base + nt * 16 + mrow) * K + ksel * 8;

  for (int k0 = 0; k0 < K; k0 += 32) {
    BF16Frag b0, b1;
    b0.u[0] = *(const u32x4*)(Wp[0] + k0);
    b0.u[1] = *(const u32x4*)(Wp[0] + k0 + 16);
    b1.u[0] = *(const u32x4*)(Wp[1] + k0);
    b1.u[1] = *(const u32x4*)(Wp[1] + k0 + 16);
    #pragma unroll
    for (int mt = 0; mt < 4; ++mt) {
      BF16Frag a;
      a.u[0] = *(const u32x4*)(Ap[mt] + k0);
      a.u[1] = *(const u32x4*)(Ap[mt] + k0 + 16);
      acc[mt][0] = __builtin_amdgcn_wmma_f32_16x16x32_bf16(
          false, a.v, false, b0.v, (short)0, acc[mt][0], false, false);
      acc[mt][1] = __builtin_amdgcn_wmma_f32_16x16x32_bf16(
          false, a.v, false, b1.v, (short)0, acc[mt][1], false, false);
    }
  }

  #pragma unroll
  for (int nt = 0; nt < 2; ++nt) {
    int col = n_base + nt * 16 + (lane & 15);
    float bv = bias0 ? bias0[col] : 0.0f;
    if (bias1) bv += bias1[col];
    #pragma unroll
    for (int mt = 0; mt < 4; ++mt) {
      int row0 = m_base + mt * 16 + (lane >> 4) * 8;
      float* cp = C + (size_t)row0 * N + col;
      #pragma unroll
      for (int r = 0; r < 8; ++r)
        cp[(size_t)r * N] = acc[mt][nt][r] + bv;
    }
  }
}

// ---------- persistent LSTM recurrence ----------
// 32 blocks x 256 threads. Block b owns H-slice [b*16, b*16+16).
// Per step: gates(64x16 per gate) = h(64x512) @ W_hh_slice^T, LDS exchange,
// activations, c update, h(t+1) slice store, grid barrier.
__global__ __launch_bounds__(256) void lstm_recur_kernel(
    const __bf16* __restrict__ Whh,    // (2048, 512) row-major, PyTorch layout
    const float*  __restrict__ xproj,  // (512, 64, 2048) f32
    __bf16* __restrict__ hs,           // (513, 64, 512): hs[0]=0, hs[t+1]=h after step t
    unsigned* bar)
{
  extern __shared__ char smem[];
  __bf16* h_lds     = (__bf16*)smem;                                  // 64*HPAD bf16 = 66560B
  float*  gates_lds = (float*)(smem + BB * HPAD * 2);                 // 4*64*GPAD f32 = 20480B
  float*  c_lds     = (float*)(smem + BB * HPAD * 2 + 4 * BB * GPAD * 4); // 64*16 f32 = 4096B

  const int tid  = threadIdx.x;
  const int lane = tid & 31;
  const int w    = tid >> 5;
  const int gate = w >> 1;          // 0..3 -> i,f,g,o
  const int mt2  = (w & 1) * 2;     // wave covers M-tiles {mt2, mt2+1}
  const int j0   = blockIdx.x * 16; // H-slice base

  for (int i = tid; i < BB * 16; i += 256) c_lds[i] = 0.0f;

  // W_hh B-fragment row pointer: output column n = gate*512 + j0 + (lane&15)
  const __bf16* Wrow = Whh + (size_t)(gate * HH + j0 + (lane & 15)) * HH + (lane >> 4) * 8;
  const int arow  = mt2 * 16 + (lane & 15);
  const int ksel8 = (lane >> 4) * 8;

  // combine-phase geometry: 4 cells per thread
  const int cm  = tid >> 2;          // batch row
  const int cjj = (tid & 3) * 4;     // column within slice (multiple of 4)

  for (int t = 0; t < TT; ++t) {
    __syncthreads();
    // ---- stage h(t) into LDS (device-coherent past the grid barrier) ----
    {
      const char* hsrc = (const char*)(hs + (size_t)t * (BB * HH));
      for (int idx = tid; idx < BB * 64; idx += 256) {
        int row = idx >> 6, chunk = idx & 63;   // 16B chunk per iteration
#ifdef USE_ASYNC_LDS
        __builtin_amdgcn_global_load_async_to_lds_b128(
            (__attribute__((address_space(1))) v4i_async*)(hsrc + row * 1024 + chunk * 16),
            (__attribute__((address_space(3))) v4i_async*)&h_lds[row * HPAD + chunk * 8],
            0, /*cpol: scope=SCOPE_DEV*/ 16);
#else
        const unsigned long long* s =
            (const unsigned long long*)(hsrc + row * 1024 + chunk * 16);
        u64x2 d;
        d.x = __hip_atomic_load(s + 0, __ATOMIC_RELAXED, __HIP_MEMORY_SCOPE_AGENT);
        d.y = __hip_atomic_load(s + 1, __ATOMIC_RELAXED, __HIP_MEMORY_SCOPE_AGENT);
        *(u64x2*)&h_lds[row * HPAD + chunk * 8] = d;
#endif
      }
#ifdef USE_ASYNC_LDS
      __builtin_amdgcn_s_wait_asynccnt(0);
#endif
    }
    __syncthreads();

    // ---- GEMM: 2 C fragments (32 batch rows) for this wave's gate column slice ----
    v8f acc0 = {}, acc1 = {};
    #pragma unroll 4
    for (int kk = 0; kk < HH; kk += 32) {
      BF16Frag bfr, a0, a1;
      bfr.u[0] = *(const u32x4*)(Wrow + kk);
      bfr.u[1] = *(const u32x4*)(Wrow + kk + 16);
      const __bf16* ap0 = &h_lds[arow * HPAD + kk + ksel8];
      a0.u[0] = *(const u32x4*)(ap0);
      a0.u[1] = *(const u32x4*)(ap0 + 16);
      const __bf16* ap1 = ap0 + 16 * HPAD;
      a1.u[0] = *(const u32x4*)(ap1);
      a1.u[1] = *(const u32x4*)(ap1 + 16);
      acc0 = __builtin_amdgcn_wmma_f32_16x16x32_bf16(
          false, a0.v, false, bfr.v, (short)0, acc0, false, false);
      acc1 = __builtin_amdgcn_wmma_f32_16x16x32_bf16(
          false, a1.v, false, bfr.v, (short)0, acc1, false, false);
    }

    // ---- scatter gate pre-activations to LDS for cross-wave combine ----
    #pragma unroll
    for (int r = 0; r < 8; ++r) {
      int m = mt2 * 16 + r + (lane >> 4) * 8;
      gates_lds[(gate * BB + m) * GPAD + (lane & 15)]      = acc0[r];
      gates_lds[(gate * BB + m + 16) * GPAD + (lane & 15)] = acc1[r];
    }
    __syncthreads();

    // ---- combine: 1024 cell updates, 4 adjacent columns per thread ----
    {
      const float* xp = xproj + (size_t)t * (BB * G4H);
      __bf16* hdst = hs + (size_t)(t + 1) * (BB * HH);
      const int col = j0 + cjj;
      f32x4 gi = *(const f32x4*)&xp[cm * G4H + 0 * HH + col];
      f32x4 gf = *(const f32x4*)&xp[cm * G4H + 1 * HH + col];
      f32x4 gg = *(const f32x4*)&xp[cm * G4H + 2 * HH + col];
      f32x4 go = *(const f32x4*)&xp[cm * G4H + 3 * HH + col];
      #pragma unroll
      for (int e = 0; e < 4; ++e) {
        gi[e] += gates_lds[(0 * BB + cm) * GPAD + cjj + e];
        gf[e] += gates_lds[(1 * BB + cm) * GPAD + cjj + e];
        gg[e] += gates_lds[(2 * BB + cm) * GPAD + cjj + e];
        go[e] += gates_lds[(3 * BB + cm) * GPAD + cjj + e];
      }
      f32x4 c = *(f32x4*)&c_lds[tid * 4];
      union { __bf16 h[4]; unsigned long long u; } pk;
      #pragma unroll
      for (int e = 0; e < 4; ++e) {
        float iv = sigmoidf_(gi[e]);
        float fv = sigmoidf_(gf[e]);
        float gv = tanhf(gg[e]);
        float ov = sigmoidf_(go[e]);
        float cv = fv * c[e] + iv * gv;
        c[e] = cv;
        pk.h[e] = (__bf16)(ov * tanhf(cv));
      }
      *(f32x4*)&c_lds[tid * 4] = c;
      __hip_atomic_store((unsigned long long*)&hdst[cm * HH + col], pk.u,
                         __ATOMIC_RELAXED, __HIP_MEMORY_SCOPE_AGENT);
    }

    // ---- grid-wide barrier across NBLK blocks ----
    __threadfence();
    __syncthreads();
    if (tid == 0) {
      unsigned ticket = __hip_atomic_fetch_add(bar, 1u, __ATOMIC_ACQ_REL,
                                               __HIP_MEMORY_SCOPE_AGENT);
      unsigned target = (ticket / NBLK + 1u) * NBLK;
      while (__hip_atomic_load(bar, __ATOMIC_ACQUIRE, __HIP_MEMORY_SCOPE_AGENT) < target)
        __builtin_amdgcn_s_sleep(2);
    }
    __syncthreads();
  }
}

// ---------- row softmax in place: (32768 rows x 1024) ----------
__global__ __launch_bounds__(256) void softmax_kernel(float* __restrict__ data) {
  __shared__ float red[8];
  const int tid  = threadIdx.x;
  const int lane = tid & 31;
  const int w    = tid >> 5;
  float* row = data + (size_t)blockIdx.x * OO;
  f32x4 v = *(const f32x4*)&row[tid * 4];
  float m = fmaxf(fmaxf(v.x, v.y), fmaxf(v.z, v.w));
  #pragma unroll
  for (int off = 16; off > 0; off >>= 1) m = fmaxf(m, __shfl_xor(m, off, 32));
  if (lane == 0) red[w] = m;
  __syncthreads();
  float rm = red[0];
  #pragma unroll
  for (int i = 1; i < 8; ++i) rm = fmaxf(rm, red[i]);
  v.x = __expf(v.x - rm); v.y = __expf(v.y - rm);
  v.z = __expf(v.z - rm); v.w = __expf(v.w - rm);
  float s = v.x + v.y + v.z + v.w;
  #pragma unroll
  for (int off = 16; off > 0; off >>= 1) s += __shfl_xor(s, off, 32);
  __syncthreads();
  if (lane == 0) red[w] = s;
  __syncthreads();
  float rs = 0.0f;
  #pragma unroll
  for (int i = 0; i < 8; ++i) rs += red[i];
  float inv = 1.0f / rs;
  v.x *= inv; v.y *= inv; v.z *= inv; v.w *= inv;
  *(f32x4*)&row[tid * 4] = v;
}

// ---------- launcher ----------
extern "C" void kernel_launch(void* const* d_in, const int* in_sizes, int n_in,
                              void* d_out, int out_size, void* d_ws, size_t ws_size,
                              hipStream_t stream) {
  const float* x     = (const float*)d_in[0];
  const float* W_ih  = (const float*)d_in[1];
  const float* W_hh  = (const float*)d_in[2];
  const float* b_ih  = (const float*)d_in[3];
  const float* b_hh  = (const float*)d_in[4];
  const float* W_out = (const float*)d_in[5];
  const float* b_out = (const float*)d_in[6];
  float* out = (float*)d_out;

  char* ws = (char*)d_ws;
  size_t off = 0;
  __bf16* x_bf  = (__bf16*)(ws + off); off += (size_t)TT * BB * II * 2;   // 16.8 MB
  __bf16* wih   = (__bf16*)(ws + off); off += (size_t)G4H * II * 2;       // 1 MB
  __bf16* whh   = (__bf16*)(ws + off); off += (size_t)G4H * HH * 2;       // 2.1 MB
  __bf16* wout  = (__bf16*)(ws + off); off += (size_t)OO * HH * 2;        // 1 MB
  float*  xproj = (float*) (ws + off); off += (size_t)TT * BB * G4H * 4;  // 268 MB
  __bf16* hs    = (__bf16*)(ws + off); off += (size_t)(TT + 1) * BB * HH * 2; // 33.6 MB
  unsigned* bar = (unsigned*)(ws + off); off += 256;

  // bf16 conversions
  cvt_bf16_kernel<<<(TT * BB * II) / 1024, 256, 0, stream>>>(x, x_bf, TT * BB * II);
  cvt_bf16_kernel<<<(G4H * II) / 1024, 256, 0, stream>>>(W_ih, wih, G4H * II);
  cvt_bf16_kernel<<<(G4H * HH) / 1024, 256, 0, stream>>>(W_hh, whh, G4H * HH);
  cvt_bf16_kernel<<<(OO * HH) / 1024, 256, 0, stream>>>(W_out, wout, OO * HH);
  init_kernel<<<64, 256, 0, stream>>>((unsigned*)hs, bar);

  // x_proj = x @ W_ih^T + b_ih + b_hh   (M=32768, N=2048, K=256)
  gemm_bf16_kernel<<<dim3(G4H / 128, (TT * BB) / 128), 256, 0, stream>>>(
      x_bf, wih, b_ih, b_hh, xproj, TT * BB, G4H, II);

  // persistent recurrence
  size_t lds_bytes = (size_t)BB * HPAD * 2 + 4 * BB * GPAD * 4 + BB * 16 * 4; // ~89 KB
  lstm_recur_kernel<<<NBLK, 256, lds_bytes, stream>>>(whh, xproj, hs, bar);

  // logits = hs @ W_out^T + b_out   (M=32768, N=1024, K=512), into d_out
  gemm_bf16_kernel<<<dim3(OO / 128, (TT * BB) / 128), 256, 0, stream>>>(
      hs + (size_t)BB * HH, wout, b_out, nullptr, out, TT * BB, OO, HH);

  softmax_kernel<<<TT * BB, 256, 0, stream>>>(out);

  (void)in_sizes; (void)n_in; (void)out_size; (void)ws_size;
}
